// UserSessionSimNet_21345987461278
// MI455X (gfx1250) — matrix operand order repaired and checked by reading, not compiled, not run on hardware
//
#include <hip/hip_runtime.h>
#include <cstdint>

#define HDIM 256
#define UDIM 20000
#define WPB  4   // waves per block in WMMA kernels (wave32)

typedef __attribute__((ext_vector_type(16))) __bf16 v16bf;
typedef __attribute__((ext_vector_type(8)))  float  v8f;

union Frag { v16bf v; unsigned u[8]; };

__device__ __forceinline__ unsigned pack_bf2(float lo, float hi) {
  union { __bf16 h[2]; unsigned u; } p;
  p.h[0] = (__bf16)lo; p.h[1] = (__bf16)hi;
  return p.u;
}

// float atomic max via sign-aware integer atomics (native global_atomic_max/min)
__device__ __forceinline__ void atomicMaxF(float* a, float v) {
  if (v >= 0.0f) atomicMax((int*)a, __float_as_int(v));
  else           atomicMin((unsigned int*)a, __float_as_uint(v));
}

// ---- CDNA5 async global->LDS copy (ASYNCcnt path), with sync fallback ----
#if __has_builtin(__builtin_amdgcn_global_load_async_to_lds_b128)
#define HAVE_ASYNC_LDS 1
typedef int v4i_b128 __attribute__((vector_size(16)));
typedef __attribute__((address_space(1))) v4i_b128* gas_v4p;
typedef __attribute__((address_space(3))) v4i_b128* las_v4p;
__device__ __forceinline__ void async_ld_b128(void* lds, const void* g) {
  __builtin_amdgcn_global_load_async_to_lds_b128((gas_v4p)g, (las_v4p)lds, 0, 0);
}
__device__ __forceinline__ void wait_async0() {
#if __has_builtin(__builtin_amdgcn_s_wait_asynccnt)
  __builtin_amdgcn_s_wait_asynccnt(0);
#else
  asm volatile("s_wait_asynccnt 0" ::: "memory");
#endif
}
#else
#define HAVE_ASYNC_LDS 0
#endif

// ---------------- init segment buffers ----------------
__global__ void k_init(float* __restrict__ segmax, float* __restrict__ denom,
                       float* __restrict__ wsum, int U) {
  long t = (long)blockIdx.x * blockDim.x + threadIdx.x;
  if (t < (long)U * HDIM) wsum[t] = 0.0f;
  if (t < U) { segmax[t] = -INFINITY; denom[t] = 0.0f; }
}

// ---------------- convert weights to bf16 (Wq/Wk/Wv transposed) ----------------
__global__ void k_cvtw(const float* __restrict__ Wq, const float* __restrict__ Wk,
                       const float* __restrict__ Wv, const float* __restrict__ W1,
                       __bf16* __restrict__ WqT, __bf16* __restrict__ WkT,
                       __bf16* __restrict__ WvT, __bf16* __restrict__ W1b) {
  int t = blockIdx.x * 256 + threadIdx.x;
  if (t < HDIM * HDIM) {
    int j = t >> 8, k = t & 255;               // WqT[j][k] = Wq[k][j]
    WqT[t] = (__bf16)Wq[k * HDIM + j];
    WkT[t] = (__bf16)Wk[k * HDIM + j];
    WvT[t] = (__bf16)Wv[k * HDIM + j];
  }
  if (t < HDIM * 2 * HDIM) W1b[t] = (__bf16)W1[t];  // already output-major for x@W1^T
}

// ---------------- Q/K/V projections + row-dot similarity + segment max ----------------
__global__ __launch_bounds__(WPB * 32) void k_proj_sim(
    const float* __restrict__ user, const float* __restrict__ sess,
    const int* __restrict__ uid,
    const __bf16* __restrict__ WqT, const __bf16* __restrict__ WkT,
    const __bf16* __restrict__ WvT,
    __bf16* __restrict__ Vb, float* __restrict__ sim,
    float* __restrict__ segmax, long N) {
  __shared__ unsigned ldsU[WPB][16 * HDIM / 2];  // 16x256 bf16, packed pairs
  __shared__ unsigned ldsS[WPB][16 * HDIM / 2];

  const int  lane   = threadIdx.x & 31;
  const int  wave   = threadIdx.x >> 5;
  const long nTiles = (N + 15) >> 4;
  const long tile   = (long)blockIdx.x * WPB + wave;
  if (tile >= nTiles) return;                    // wave-uniform: EXEC stays all-ones

  const long base = tile * 16 * HDIM;            // tile is 4096 contiguous floats
  const long totF = N * HDIM;
  for (int it = 0; it < 32; ++it) {
    long fo = base + (long)(it * 32 + lane) * 4;
    float4 xu = make_float4(0.f, 0.f, 0.f, 0.f), xs = xu;
    if (fo + 3 < totF) {
      xu = *(const float4*)(user + fo);
      xs = *(const float4*)(sess + fo);
    }
    int d = (it * 32 + lane) * 2;
    ldsU[wave][d]     = pack_bf2(xu.x, xu.y);
    ldsU[wave][d + 1] = pack_bf2(xu.z, xu.w);
    ldsS[wave][d]     = pack_bf2(xs.x, xs.y);
    ldsS[wave][d + 1] = pack_bf2(xs.z, xs.w);
  }
  asm volatile("s_wait_dscnt 0" ::: "memory");   // DS writes visible before lane-crossing reads

  const int row = lane & 15;
  const int hlf = lane >> 4;
  float sdot[8];
#pragma unroll
  for (int v = 0; v < 8; ++v) sdot[v] = 0.f;

  for (int ct = 0; ct < 16; ++ct) {
    const int col = ct * 16 + (lane & 15);
    v8f cQ = {}; v8f cK = {}; v8f cV = {};
#pragma unroll
    for (int kc = 0; kc < 8; ++kc) {
      const int kb = kc * 32;
      Frag aU, aS, bQ, bK, bV;
#pragma unroll
      for (int v = 0; v < 8; ++v) {
        // A 16x32 bf16 layout: lanes0-15 K=0..7/16..23, lanes16-31 K=8..15/24..31
        int k0 = (v < 4) ? (hlf * 8 + 2 * v) : (16 + hlf * 8 + 2 * (v - 4));
        int ai = row * 128 + ((kb + k0) >> 1);
        aU.u[v] = ldsU[wave][ai];
        aS.u[v] = ldsS[wave][ai];
        // B 32x16 bf16 layout: lanes0-15 K=0..7/16..23, lanes16-31 K=8..15/24..31; pairs contiguous in W*T rows
        int k  = kb + k0;
        int wi = col * HDIM + k;
        bQ.u[v] = *(const unsigned*)(WqT + wi);
        bK.u[v] = *(const unsigned*)(WkT + wi);
        bV.u[v] = *(const unsigned*)(WvT + wi);
      }
      cQ = __builtin_amdgcn_wmma_f32_16x16x32_bf16(false, aU.v, false, bQ.v, (short)0, cQ, false, false);
      cK = __builtin_amdgcn_wmma_f32_16x16x32_bf16(false, aS.v, false, bK.v, (short)0, cK, false, false);
      cV = __builtin_amdgcn_wmma_f32_16x16x32_bf16(false, aU.v, false, bV.v, (short)0, cV, false, false);
    }
#pragma unroll
    for (int v = 0; v < 8; ++v) {
      sdot[v] += cQ[v] * cK[v];                          // diag(Q K^T) partial over this col-tile
      long r = tile * 16 + hlf * 8 + v;                  // C layout: VGPR v -> M=v / v+8
      if (r < N) Vb[r * HDIM + col] = (__bf16)cV[v];
    }
  }
  // reduce over N within each 16-lane half -> full row dot products
#pragma unroll
  for (int v = 0; v < 8; ++v) {
    float s = sdot[v];
    s += __shfl_xor(s, 1, 32);
    s += __shfl_xor(s, 2, 32);
    s += __shfl_xor(s, 4, 32);
    s += __shfl_xor(s, 8, 32);
    sdot[v] = s;
  }
  if ((lane & 15) == 0) {
#pragma unroll
    for (int v = 0; v < 8; ++v) {
      long r = tile * 16 + hlf * 8 + v;
      if (r < N) {
        sim[r] = sdot[v];
        atomicMaxF(&segmax[uid[r]], sdot[v]);
      }
    }
  }
}

// ---------------- exp + segment sums (denominator and unnormalized weighted sum) ----------------
__global__ void k_expsum(const float* __restrict__ user, const int* __restrict__ uid,
                         const float* __restrict__ sim, const float* __restrict__ segmax,
                         float* __restrict__ denom, float* __restrict__ wsum, long N) {
  const int  t = threadIdx.x & 63;                 // 64 threads per row
  const long i = (long)blockIdx.x * 4 + (threadIdx.x >> 6);
  if (i >= N) return;
  const int   u = uid[i];
  const float e = __expf(sim[i] - segmax[u]);
  if (t == 0) unsafeAtomicAdd(&denom[u], e);
  const float* ue = user + i * HDIM;
  float* w = wsum + (long)u * HDIM;
  const int h0 = t * 4;
#pragma unroll
  for (int j = 0; j < 4; ++j) unsafeAtomicAdd(&w[h0 + j], ue[h0 + j] * e);
}

// ---------------- normalize weighted sums, emit bf16 ----------------
__global__ void k_norm(const float* __restrict__ wsum, const float* __restrict__ denom,
                       __bf16* __restrict__ wsumb, int U) {
  long t = (long)blockIdx.x * blockDim.x + threadIdx.x;
  if (t >= (long)U * HDIM) return;
  float d = denom[t >> 8];
  wsumb[t] = (__bf16)(d > 0.f ? wsum[t] / d : 0.f);
}

// ---------------- FFN: relu([V | wsum[uid]] @ W1^T + b1) ----------------
__global__ __launch_bounds__(WPB * 32) void k_ffn(
    const __bf16* __restrict__ Vb, const __bf16* __restrict__ wsumb,
    const int* __restrict__ uid, const __bf16* __restrict__ W1b,
    const float* __restrict__ b1, float* __restrict__ out, long N) {
  __shared__ unsigned ldsA[WPB][16 * 512 / 2];     // 16x512 bf16 concat tile

  const int  lane   = threadIdx.x & 31;
  const int  wave   = threadIdx.x >> 5;
  const long nTiles = (N + 15) >> 4;
  const long tile   = (long)blockIdx.x * WPB + wave;
  if (tile >= nTiles) return;

  const unsigned* VbU = (const unsigned*)Vb;       // bf16 rows viewed as 128 dwords
  const unsigned* WsU = (const unsigned*)wsumb;

#if HAVE_ASYNC_LDS
  // stage concat tile with CDNA5 async global->LDS B128 copies (no VGPR round-trip)
  for (int it = 0; it < 32; ++it) {
    int  o    = (it * 32 + lane) * 4;              // dword index in [0,4096)
    int  r    = o >> 8;                            // row 0..15 (256 dwords per row)
    int  c    = o & 255;
    long grow = tile * 16 + r;
    if (grow >= N) grow = N - 1;                   // clamp tail (never OOB; N%16==0 in practice)
    const unsigned* src = (c < 128) ? (VbU + grow * 128 + c)
                                    : (WsU + (long)uid[grow] * 128 + (c - 128));
    async_ld_b128(&ldsA[wave][o], src);
  }
  wait_async0();                                   // async engine done writing LDS
#else
  for (int it = 0; it < 32; ++it) {
    int  o    = (it * 32 + lane) * 4;
    int  r    = o >> 8;
    int  c    = o & 255;
    long grow = tile * 16 + r;
    uint4 x   = make_uint4(0u, 0u, 0u, 0u);
    if (grow < N) {
      if (c < 128) x = *(const uint4*)(VbU + grow * 128 + c);
      else         x = *(const uint4*)(WsU + (long)uid[grow] * 128 + (c - 128));
    }
    *(uint4*)&ldsA[wave][o] = x;
  }
  asm volatile("s_wait_dscnt 0" ::: "memory");
#endif

  const int row = lane & 15;
  const int hlf = lane >> 4;

  for (int ct = 0; ct < 16; ++ct) {
    const int col = ct * 16 + (lane & 15);
    v8f c = {};
#pragma unroll
    for (int kc = 0; kc < 16; ++kc) {
      const int kb = kc * 32;
      Frag a, b;
#pragma unroll
      for (int v = 0; v < 8; ++v) {
        int k0 = (v < 4) ? (hlf * 8 + 2 * v) : (16 + hlf * 8 + 2 * (v - 4));
        a.u[v] = ldsA[wave][row * 256 + ((kb + k0) >> 1)];
        int k  = kb + k0;
        b.u[v] = *(const unsigned*)(W1b + col * 512 + k);   // W1[j][k], pairs contiguous
      }
      c = __builtin_amdgcn_wmma_f32_16x16x32_bf16(false, a.v, false, b.v, (short)0, c, false, false);
    }
    const float bb = b1[col];
#pragma unroll
    for (int v = 0; v < 8; ++v) {
      long r = tile * 16 + hlf * 8 + v;
      if (r < N) out[r * HDIM + col] = fmaxf(c[v] + bb, 0.f);
    }
  }
}

// ---------------- launch ----------------
extern "C" void kernel_launch(void* const* d_in, const int* in_sizes, int n_in,
                              void* d_out, int out_size, void* d_ws, size_t ws_size,
                              hipStream_t stream) {
  const float* sess = (const float*)d_in[0];
  const float* user = (const float*)d_in[1];
  const int*   uid  = (const int*)d_in[2];
  const float* Wq   = (const float*)d_in[3];
  const float* Wk   = (const float*)d_in[4];
  const float* Wv   = (const float*)d_in[5];
  const float* W1   = (const float*)d_in[6];
  const float* b1   = (const float*)d_in[7];
  float* out = (float*)d_out;
  const long N = (long)in_sizes[0] / HDIM;
  const int  U = UDIM;
  (void)n_in; (void)out_size; (void)ws_size;

  char*  ws  = (char*)d_ws;
  size_t off = 0;
  auto carve = [&](size_t bytes) -> void* {
    void* p = ws + off;
    off += (bytes + 255) & ~(size_t)255;
    return p;
  };
  __bf16* WqT    = (__bf16*)carve((size_t)HDIM * HDIM * 2);
  __bf16* WkT    = (__bf16*)carve((size_t)HDIM * HDIM * 2);
  __bf16* WvT    = (__bf16*)carve((size_t)HDIM * HDIM * 2);
  __bf16* W1b    = (__bf16*)carve((size_t)HDIM * 2 * HDIM * 2);
  float*  sim    = (float*)carve((size_t)N * 4);
  float*  segmax = (float*)carve((size_t)U * 4);
  float*  denom  = (float*)carve((size_t)U * 4);
  float*  wsum   = (float*)carve((size_t)U * HDIM * 4);
  __bf16* wsumb  = (__bf16*)carve((size_t)U * HDIM * 2);
  __bf16* Vb     = (__bf16*)carve((size_t)N * HDIM * 2);

  const long segTot = (long)U * HDIM;
  k_init<<<(segTot + 255) / 256, 256, 0, stream>>>(segmax, denom, wsum, U);
  k_cvtw<<<(HDIM * 2 * HDIM + 255) / 256, 256, 0, stream>>>(Wq, Wk, Wv, W1, WqT, WkT, WvT, W1b);

  const long nTiles = (N + 15) / 16;
  const long blocks = (nTiles + WPB - 1) / WPB;
  k_proj_sim<<<blocks, WPB * 32, 0, stream>>>(user, sess, uid, WqT, WkT, WvT, Vb, sim, segmax, N);
  k_expsum<<<(N + 3) / 4, 256, 0, stream>>>(user, uid, sim, segmax, denom, wsum, N);
  k_norm<<<(segTot + 255) / 256, 256, 0, stream>>>(wsum, denom, wsumb, U);
  k_ffn<<<blocks, WPB * 32, 0, stream>>>(Vb, wsumb, uid, W1b, b1, out, N);
}
